// SurfaceToVolumeModel_45938970198404
// MI455X (gfx1250) — matrix operand order repaired and checked by reading, not compiled, not run on hardware
//
#include <hip/hip_runtime.h>
#include <cstddef>

// ---------------- problem constants ----------------
#define BB   2
#define NN   4096
#define TT   8192
#define KNB  20
#define KLOC 8
#define MDEC (BB*TT)     // 16384 decoder rows
#define NIP  800         // 771 padded to multiple of 32

typedef __attribute__((ext_vector_type(16))) __bf16 v16bf;
typedef __attribute__((ext_vector_type(8)))  __bf16 v8bf;
typedef __attribute__((ext_vector_type(8)))  float  v8f;

__device__ inline v16bf cat16(v8bf lo, v8bf hi) {
  return __builtin_shufflevector(lo, hi, 0,1,2,3,4,5,6,7,8,9,10,11,12,13,14,15);
}

// =====================================================================
// elementwise helpers
// =====================================================================
__global__ void sqnorm_kernel(const float* __restrict__ X, int ldx, int xoff, int C,
                              float* __restrict__ sq, int total) {
  int i = blockIdx.x * blockDim.x + threadIdx.x;
  if (i >= total) return;
  const float* p = X + (size_t)i * ldx + xoff;
  float s = 0.f;
  for (int c = 0; c < C; ++c) { float v = p[c]; s += v * v; }
  sq[i] = s;
}

__global__ void copy_kernel(const float* __restrict__ s, float* __restrict__ d, size_t n) {
  size_t i = (size_t)blockIdx.x * blockDim.x + threadIdx.x;
  if (i < n) d[i] = s[i];
}

__global__ void add_kernel(float* __restrict__ d, const float* __restrict__ s, size_t n) {
  size_t i = (size_t)blockIdx.x * blockDim.x + threadIdx.x;
  if (i < n) d[i] += s[i];
}

__global__ void sub_kernel(const float* __restrict__ a, const float* __restrict__ b,
                           float* __restrict__ o, size_t n) {
  size_t i = (size_t)blockIdx.x * blockDim.x + threadIdx.x;
  if (i < n) o[i] = a[i] - b[i];
}

// =====================================================================
// pack weights (O x Kd, f32) -> bf16 fragments in B-matrix lane order:
//   P[((ot*chunks + kc)*32 + lane)*16 + e]  = W[o][kc*32 + (e < 8 ? half*8 + e : 16 + half*8 + (e - 8))]
// =====================================================================
__global__ void pack_w_kernel(const float* __restrict__ W, __bf16* __restrict__ P,
                              int O, int Kd, int Kp) {
  int chunks = Kp >> 5;
  int OTt = (O + 15) >> 4;
  size_t total = (size_t)OTt * chunks * 32 * 16;
  size_t i = (size_t)blockIdx.x * blockDim.x + threadIdx.x;
  if (i >= total) return;
  int e    = (int)(i & 15);
  int lane = (int)((i >> 4) & 31);
  size_t rest = i >> 9;
  int kc = (int)(rest % chunks);
  int ot = (int)(rest / chunks);
  int col = lane & 15, half = lane >> 4;
  int o  = ot * 16 + col;
  int kg = kc * 32 + (e < 8 ? half * 8 + e : 16 + half * 8 + (e - 8));
  float v = (o < O && kg < Kd) ? W[(size_t)o * Kd + kg] : 0.f;
  P[i] = (__bf16)v;
}

// =====================================================================
// KNN: one wave per query point, fused distance + top-KNB.
// =====================================================================
__global__ void knn_kernel(const float* __restrict__ X, int ldx, int xoff, int C,
                           const float* __restrict__ sq, int* __restrict__ out_idx) {
  __shared__ float xi[256];
  __shared__ float cd[32 * KNB];
  __shared__ int   ci[32 * KNB];
  __shared__ int   heads[32];

  int b = blockIdx.x / NN;
  int i = blockIdx.x % NN;
  int lane = threadIdx.x;
  const float* Xb = X + (size_t)b * NN * ldx + xoff;

  for (int c = lane; c < C; c += 32) xi[c] = Xb[(size_t)i * ldx + c];
  __syncthreads();

  float sqi = sq[b * NN + i];
  float* md = cd + lane * KNB;
  int*   mi = ci + lane * KNB;
  int cnt = 0;
  for (int j = lane; j < NN; j += 32) {
    const float* xj = Xb + (size_t)j * ldx;
    float dot = 0.f;
    for (int c = 0; c < C; ++c) dot += xi[c] * xj[c];
    float d = sqi + sq[b * NN + j] - 2.f * dot;
    if (cnt < KNB) {
      int p = cnt++;
      while (p > 0 && md[p - 1] > d) { md[p] = md[p - 1]; mi[p] = mi[p - 1]; --p; }
      md[p] = d; mi[p] = j;
    } else if (d < md[KNB - 1]) {
      int p = KNB - 1;
      while (p > 0 && md[p - 1] > d) { md[p] = md[p - 1]; mi[p] = mi[p - 1]; --p; }
      md[p] = d; mi[p] = j;
    }
  }
  __syncthreads();
  if (lane == 0) {
    for (int l = 0; l < 32; ++l) heads[l] = 0;
    int* oi = out_idx + ((size_t)b * NN + i) * KNB;
    for (int t = 0; t < KNB; ++t) {
      float best = 3.4e38f; int bl = 0;
      for (int l = 0; l < 32; ++l) {
        int h = heads[l];
        if (h < KNB && cd[l * KNB + h] < best) { best = cd[l * KNB + h]; bl = l; }
      }
      oi[t] = ci[bl * KNB + heads[bl]];
      heads[bl]++;
    }
  }
}

// =====================================================================
// Encoder edge-conv / projection GEMM, WMMA bf16, packed weights,
// NO output tiles per wave (amortizes LDS staging + A traffic).
//   PASS 0: per-(batch,channel) sum/sumsq;  PASS 1: GN + lrelu + max_k.
// =====================================================================
template <int PASS, int NO>
__global__ void enc_gemm_kernel(const float* __restrict__ X, int ldx, int xoff, int Cin,
                                const int* __restrict__ nidx, int knb,
                                const __bf16* __restrict__ Wp,   // packed fragments
                                const float* __restrict__ gamma, const float* __restrict__ beta,
                                float* __restrict__ chsum, float* __restrict__ chsq,
                                float* __restrict__ out, int ldo, int ooff,
                                int O, int groups) {
  __shared__ v8bf  lsA8[16 * 512 / 8];
  __shared__ float lsX[16 * 448];
  __shared__ int   jrow[16];
  __bf16* lsA = (__bf16*)lsA8;

  const int Kdim   = nidx ? (2 * Cin) : Cin;
  const int KdimP  = (Kdim + 31) & ~31;
  const int chunks = KdimP >> 5;

  int OG = O / (16 * NO);            // O-tile groups
  int og = blockIdx.x % OG;
  int t2 = blockIdx.x / OG;
  int nt = t2 % (NN >> 4);
  int b  = t2 / (NN >> 4);

  int lane = threadIdx.x;
  int row  = lane & 15;
  int half = lane >> 4;

  const float* Xb = X + (size_t)b * NN * ldx + xoff;

  // stage center features (reused for all kk)
  for (int e = lane; e < 16 * Cin; e += 32) {
    int r = e / Cin, c = e % Cin;
    lsX[e] = Xb[(size_t)(nt * 16 + r) * ldx + c];
  }
  __syncthreads();
  // stage kk-invariant part of A (center half + zero pad)
  for (int e = lane; e < 16 * KdimP; e += 32) {
    int r = e / KdimP, c = e % KdimP;
    if (nidx) {
      if (c < Cin) continue;   // neighbor-diff half, staged per kk
      float v = (c < Kdim) ? lsX[r * Cin + (c - Cin)] : 0.f;
      lsA[e] = (__bf16)v;
    } else {
      lsA[e] = (__bf16)((c < Kdim) ? lsX[r * Cin + c] : 0.f);
    }
  }

  int   oj[NO];
  float mean[NO], inv[NO], ga[NO], be[NO];
#pragma unroll
  for (int j = 0; j < NO; ++j) {
    oj[j] = (og * NO + j) * 16 + row;
    mean[j] = 0.f; inv[j] = 1.f; ga[j] = 1.f; be[j] = 0.f;
  }
  if (PASS == 1) {
    int Cg = O / groups;
    float cntf = (float)NN * (float)knb * (float)Cg;
#pragma unroll
    for (int j = 0; j < NO; ++j) {
      int g = oj[j] / Cg;
      float gs = 0.f, gq = 0.f;
      for (int c = 0; c < Cg; ++c) {
        gs += chsum[b * O + g * Cg + c];
        gq += chsq [b * O + g * Cg + c];
      }
      mean[j] = gs / cntf;
      float var = gq / cntf - mean[j] * mean[j];
      inv[j] = rsqrtf(var + 1e-5f);
      ga[j] = gamma[oj[j]]; be[j] = beta[oj[j]];
    }
  }

  float s1[NO], s2[NO];
  v8f acc[NO];
#pragma unroll
  for (int j = 0; j < NO; ++j) {
    s1[j] = 0.f; s2[j] = 0.f;
#pragma unroll
    for (int r = 0; r < 8; ++r) acc[j][r] = -3.4e38f;
  }

  const __bf16* wbase = Wp + ((size_t)(og * NO) * chunks * 32 + lane) * 16;

  for (int kk = 0; kk < knb; ++kk) {
    if (nidx) {
      __syncthreads();
      if (lane < 16) jrow[lane] = nidx[((size_t)b * NN + nt * 16 + lane) * knb + kk];
      __syncthreads();
      for (int e = lane; e < 16 * Cin; e += 32) {
        int r = e / Cin, c = e % Cin;
        lsA[r * KdimP + c] = (__bf16)(Xb[(size_t)jrow[r] * ldx + c] - lsX[e]);
      }
    }
    __syncthreads();

    v8f c8[NO];
#pragma unroll
    for (int j = 0; j < NO; ++j)
#pragma unroll
      for (int r = 0; r < 8; ++r) c8[j][r] = 0.f;

    for (int kc = 0; kc < chunks; ++kc) {
      const v8bf* rp = lsA8 + ((row * KdimP + kc * 32 + half * 8) >> 3);
      v16bf a = cat16(rp[0], rp[2]);
#pragma unroll
      for (int j = 0; j < NO; ++j) {
        v16bf bw = *(const v16bf*)(wbase + ((size_t)j * chunks + kc) * 512);
        c8[j] = __builtin_amdgcn_wmma_f32_16x16x32_bf16(false, a, false, bw,
                                                        (short)0, c8[j], false, false);
      }
    }

    if (PASS == 0) {
#pragma unroll
      for (int j = 0; j < NO; ++j)
#pragma unroll
        for (int r = 0; r < 8; ++r) { s1[j] += c8[j][r]; s2[j] += c8[j][r] * c8[j][r]; }
    } else {
#pragma unroll
      for (int j = 0; j < NO; ++j)
#pragma unroll
        for (int r = 0; r < 8; ++r) {
          float y = (c8[j][r] - mean[j]) * inv[j] * ga[j] + be[j];
          y = (y > 0.f) ? y : 0.2f * y;
          acc[j][r] = fmaxf(acc[j][r], y);
        }
    }
  }

  if (PASS == 0) {
#pragma unroll
    for (int j = 0; j < NO; ++j) {
      atomicAdd(&chsum[b * O + oj[j]], s1[j]);
      atomicAdd(&chsq [b * O + oj[j]], s2[j]);
    }
  } else {
#pragma unroll
    for (int j = 0; j < NO; ++j)
#pragma unroll
      for (int r = 0; r < 8; ++r) {
        int n = nt * 16 + r + (half ? 8 : 0);
        out[((size_t)b * NN + n) * ldo + ooff + oj[j]] = acc[j][r];
      }
  }
}

// =====================================================================
// global_feat = concat(max_n g, mean_n g)
// =====================================================================
__global__ void greduce_kernel(const float* __restrict__ g, float* __restrict__ gf) {
  __shared__ float smax[256], ssum[256];
  int b = blockIdx.x >> 8;
  int c = blockIdx.x & 255;
  int t = threadIdx.x;
  float mx = -3.4e38f, sm = 0.f;
  for (int n = t; n < NN; n += 256) {
    float v = g[((size_t)b * NN + n) * 256 + c];
    mx = fmaxf(mx, v); sm += v;
  }
  smax[t] = mx; ssum[t] = sm;
  __syncthreads();
  for (int s = 128; s > 0; s >>= 1) {
    if (t < s) { smax[t] = fmaxf(smax[t], smax[t + s]); ssum[t] += ssum[t + s]; }
    __syncthreads();
  }
  if (t == 0) {
    gf[b * 512 + c]       = smax[0];
    gf[b * 512 + 256 + c] = ssum[0] / (float)NN;
  }
}

// =====================================================================
// decoder local features: wave per template point, fused top-8 +
// inverse-distance-weighted gather of point_feat (256ch).
// =====================================================================
__global__ void local_feat_kernel(const float* __restrict__ pos,
                                  const float* __restrict__ surf,   // ld 6
                                  const float* __restrict__ ssq,
                                  const float* __restrict__ pf,
                                  float* __restrict__ local) {
  __shared__ float cd[32 * KLOC];
  __shared__ int   ci[32 * KLOC];
  __shared__ float td[KLOC];
  __shared__ int   ti[KLOC];
  __shared__ int   heads[32];

  int b = blockIdx.x / TT;
  int t = blockIdx.x % TT;
  int lane = threadIdx.x;

  const float* p = pos + ((size_t)b * TT + t) * 3;
  float px = p[0], py = p[1], pz = p[2];
  float psq = px * px + py * py + pz * pz;
  const float* Sb = surf + (size_t)b * NN * 6;

  float* md = cd + lane * KLOC;
  int*   mi = ci + lane * KLOC;
  int cnt = 0;
  for (int j = lane; j < NN; j += 32) {
    const float* s = Sb + (size_t)j * 6;
    float d2 = psq + ssq[b * NN + j] - 2.f * (px * s[0] + py * s[1] + pz * s[2]);
    float d = sqrtf(fmaxf(d2, 1e-12f));
    if (cnt < KLOC) {
      int q = cnt++;
      while (q > 0 && md[q - 1] > d) { md[q] = md[q - 1]; mi[q] = mi[q - 1]; --q; }
      md[q] = d; mi[q] = j;
    } else if (d < md[KLOC - 1]) {
      int q = KLOC - 1;
      while (q > 0 && md[q - 1] > d) { md[q] = md[q - 1]; mi[q] = mi[q - 1]; --q; }
      md[q] = d; mi[q] = j;
    }
  }
  __syncthreads();
  if (lane == 0) {
    for (int l = 0; l < 32; ++l) heads[l] = 0;
    for (int k = 0; k < KLOC; ++k) {
      float best = 3.4e38f; int bl = 0;
      for (int l = 0; l < 32; ++l) {
        int h = heads[l];
        if (cd[l * KLOC + h] < best) { best = cd[l * KLOC + h]; bl = l; }
      }
      td[k] = best; ti[k] = ci[bl * KLOC + heads[bl]];
      heads[bl]++;
    }
  }
  __syncthreads();

  float w[KLOC]; float wsum = 0.f;
#pragma unroll
  for (int k = 0; k < KLOC; ++k) { w[k] = 1.f / (td[k] + 1e-8f); wsum += w[k]; }
  float invs = 1.f / wsum;
  for (int c = lane; c < 256; c += 32) {
    float v = 0.f;
#pragma unroll
    for (int k = 0; k < KLOC; ++k)
      v += w[k] * invs * pf[((size_t)b * NN + ti[k]) * 256 + c];
    local[((size_t)b * TT + t) * 256 + c] = v;
  }
}

// =====================================================================
// ni = concat(pos(3), local(256), gf(512)) -> bf16 (B,T,NIP) zero-padded
// =====================================================================
__global__ void build_ni_kernel(const float* __restrict__ pos,
                                const float* __restrict__ local,
                                const float* __restrict__ gf,
                                __bf16* __restrict__ ni) {
  size_t i = (size_t)blockIdx.x * blockDim.x + threadIdx.x;
  size_t total = (size_t)BB * TT * NIP;
  if (i >= total) return;
  int c = (int)(i % NIP);
  size_t bt = i / NIP;
  int b = (int)(bt / TT);
  float v = 0.f;
  if (c < 3)        v = pos[bt * 3 + c];
  else if (c < 259) v = local[bt * 256 + (c - 3)];
  else if (c < 771) v = gf[b * 512 + (c - 259)];
  ni[i] = (__bf16)v;
}

// =====================================================================
// decoder GEMM: out = [acc +] act(A_bf16 @ Wpacked + bias) * scale
// NO output tiles per wave; A fragment loaded once, feeds NO WMMAs.
// act: 0=none 1=relu 2=sigmoid.
// =====================================================================
template <typename OutT, int NO>
__global__ void dec_gemm_kernel(const __bf16* __restrict__ A, int lda,
                                const __bf16* __restrict__ Wp, const float* __restrict__ bias,
                                OutT* __restrict__ out, int ldo,
                                int Kp, int O,
                                int act, int accumulate, float scale) {
  int OG = (O + 16 * NO - 1) / (16 * NO);
  int og = blockIdx.x % OG;
  int mt = blockIdx.x / OG;
  int lane = threadIdx.x;
  int row  = lane & 15;
  int half = lane >> 4;
  int chunks = Kp >> 5;

  v8f c8[NO];
#pragma unroll
  for (int j = 0; j < NO; ++j)
#pragma unroll
    for (int r = 0; r < 8; ++r) c8[j][r] = 0.f;

  const __bf16* arow  = A  + (size_t)(mt * 16 + row) * lda + half * 8;
  const __bf16* wbase = Wp + ((size_t)(og * NO) * chunks * 32 + lane) * 16;

  for (int kc = 0; kc < chunks; ++kc) {
    __builtin_prefetch(arow + (kc + 1) * 32, 0, 1);
    v8bf lo = *(const v8bf*)(arow + kc * 32);
    v8bf hi = *(const v8bf*)(arow + kc * 32 + 16);
    v16bf a = cat16(lo, hi);
#pragma unroll
    for (int j = 0; j < NO; ++j) {
      v16bf bw = *(const v16bf*)(wbase + ((size_t)j * chunks + kc) * 512);
      c8[j] = __builtin_amdgcn_wmma_f32_16x16x32_bf16(false, a, false, bw,
                                                      (short)0, c8[j], false, false);
    }
  }

#pragma unroll
  for (int j = 0; j < NO; ++j) {
    int o = (og * NO + j) * 16 + row;
    if (o < O) {
      float bv = bias ? bias[o] : 0.f;
#pragma unroll
      for (int r = 0; r < 8; ++r) {
        int m = mt * 16 + r + (half ? 8 : 0);
        float v = c8[j][r] + bv;
        if (act == 1)      v = fmaxf(v, 0.f);
        else if (act == 2) v = 1.f / (1.f + __expf(-v));
        v *= scale;
        OutT* po = &out[(size_t)m * ldo + o];
        if (accumulate) v += (float)*po;
        *po = (OutT)v;
      }
    }
  }
}

// =====================================================================
// host launch
// =====================================================================
extern "C" void kernel_launch(void* const* d_in, const int* in_sizes, int n_in,
                              void* d_out, int out_size, void* d_ws, size_t ws_size,
                              hipStream_t stream) {
  const float* surface = (const float*)d_in[0];   // (2,4096,6)
  const float* templ   = (const float*)d_in[1];   // (2,8192,3)
  const float* W1 = (const float*)d_in[2],  *g1 = (const float*)d_in[3],  *b1 = (const float*)d_in[4];
  const float* W2 = (const float*)d_in[5],  *g2 = (const float*)d_in[6],  *b2 = (const float*)d_in[7];
  const float* W3 = (const float*)d_in[8],  *g3 = (const float*)d_in[9],  *b3 = (const float*)d_in[10];
  const float* Wp = (const float*)d_in[11], *gp = (const float*)d_in[12], *bp = (const float*)d_in[13];
  const float* Wg = (const float*)d_in[14], *gg = (const float*)d_in[15], *bg = (const float*)d_in[16];
  const float* fc1W = (const float*)d_in[17], *fc1b = (const float*)d_in[18];
  const float* fc2W = (const float*)d_in[19], *fc2b = (const float*)d_in[20];
  const float* skW  = (const float*)d_in[21], *skb  = (const float*)d_in[22];
  const float* outW = (const float*)d_in[23], *outb = (const float*)d_in[24];
  const float* m1W  = (const float*)d_in[25], *m1b  = (const float*)d_in[26];
  const float* m2W  = (const float*)d_in[27], *m2b  = (const float*)d_in[28];
  const float* m3W  = (const float*)d_in[29], *m3b  = (const float*)d_in[30];

  char* ws = (char*)d_ws;
  size_t off = 0;
  auto wsalloc = [&](size_t bytes) -> void* {
    void* p = ws + off;
    off += (bytes + 255) & ~(size_t)255;
    return p;
  };

  // f32 buffers
  float* sqb   = (float*)wsalloc(sizeof(float) * BB * NN);
  int*   idxb  = (int*)  wsalloc(sizeof(int)   * BB * NN * KNB);
  float* cat   = (float*)wsalloc(sizeof(float) * (size_t)BB * NN * 448);
  float* chsum = (float*)wsalloc(sizeof(float) * BB * 256);
  float* chsq  = (float*)wsalloc(sizeof(float) * BB * 256);
  float* pf    = (float*)wsalloc(sizeof(float) * (size_t)BB * NN * 256);
  float* gb    = (float*)wsalloc(sizeof(float) * (size_t)BB * NN * 256);
  float* gf    = (float*)wsalloc(sizeof(float) * BB * 512);
  float* ssq   = (float*)wsalloc(sizeof(float) * BB * NN);
  float* pos   = (float*)wsalloc(sizeof(float) * (size_t)BB * TT * 3);
  float* local = (float*)wsalloc(sizeof(float) * (size_t)BB * TT * 256);
  float* disp  = (float*)wsalloc(sizeof(float) * (size_t)BB * TT * 3);
  // bf16 activations (K-padded)
  __bf16* ni  = (__bf16*)wsalloc(sizeof(__bf16) * (size_t)BB * TT * NIP);
  __bf16* h1  = (__bf16*)wsalloc(sizeof(__bf16) * (size_t)BB * TT * 256);
  __bf16* h2  = (__bf16*)wsalloc(sizeof(__bf16) * (size_t)BB * TT * 256);
  __bf16* mb1 = (__bf16*)wsalloc(sizeof(__bf16) * (size_t)BB * TT * 128);
  __bf16* mb2 = (__bf16*)wsalloc(sizeof(__bf16) * (size_t)BB * TT * 64);

  // packed bf16 weights
  auto psz = [](int O, int Kp) -> size_t { return (size_t)((O + 15) / 16) * (Kp / 32) * 512; };
  __bf16* pW1   = (__bf16*)wsalloc(sizeof(__bf16) * psz(64, 32));
  __bf16* pW2   = (__bf16*)wsalloc(sizeof(__bf16) * psz(128, 128));
  __bf16* pW3   = (__bf16*)wsalloc(sizeof(__bf16) * psz(256, 256));
  __bf16* pWp   = (__bf16*)wsalloc(sizeof(__bf16) * psz(256, 448));
  __bf16* pWg   = (__bf16*)wsalloc(sizeof(__bf16) * psz(256, 448));
  __bf16* pFc1  = (__bf16*)wsalloc(sizeof(__bf16) * psz(256, NIP));
  __bf16* pSk   = (__bf16*)wsalloc(sizeof(__bf16) * psz(256, NIP));
  __bf16* pM1   = (__bf16*)wsalloc(sizeof(__bf16) * psz(128, NIP));
  __bf16* pFc2  = (__bf16*)wsalloc(sizeof(__bf16) * psz(256, 256));
  __bf16* pM2   = (__bf16*)wsalloc(sizeof(__bf16) * psz(64, 128));
  __bf16* pOut  = (__bf16*)wsalloc(sizeof(__bf16) * psz(3, 256));
  __bf16* pM3   = (__bf16*)wsalloc(sizeof(__bf16) * psz(1, 64));

  float* out_delta = (float*)d_out;                        // (B,T,3)
  float* out_mat   = (float*)d_out + (size_t)BB * TT * 3;  // (B,T)

  auto pack = [&](const float* W, __bf16* P, int O, int Kd, int Kp) {
    size_t tot = psz(O, Kp);
    pack_w_kernel<<<(unsigned)((tot + 255) / 256), 256, 0, stream>>>(W, P, O, Kd, Kp);
  };
  pack(W1, pW1, 64, 12, 32);
  pack(W2, pW2, 128, 128, 128);
  pack(W3, pW3, 256, 256, 256);
  pack(Wp, pWp, 256, 448, 448);
  pack(Wg, pWg, 256, 448, 448);
  pack(fc1W, pFc1, 256, 771, NIP);
  pack(skW,  pSk,  256, 771, NIP);
  pack(m1W,  pM1,  128, 771, NIP);
  pack(fc2W, pFc2, 256, 256, 256);
  pack(m2W,  pM2,  64, 128, 128);
  pack(outW, pOut, 3, 256, 256);
  pack(m3W,  pM3,  1, 64, 64);

  const int NTILE = NN / 16;
  const int EV = BB * NN;

  auto run_edge_layer = [&](const float* X, int ldx, int xoff, int Cin,
                            const __bf16* Wpk, const float* ga, const float* be,
                            float* out, int ooff, int O, int groups) {
    sqnorm_kernel<<<(EV + 255) / 256, 256, 0, stream>>>(X, ldx, xoff, Cin, sqb, EV);
    knn_kernel<<<BB * NN, 32, 0, stream>>>(X, ldx, xoff, Cin, sqb, idxb);
    hipMemsetAsync(chsum, 0, sizeof(float) * BB * 256, stream);
    hipMemsetAsync(chsq,  0, sizeof(float) * BB * 256, stream);
    int grid = BB * NTILE * (O / 64);   // NO=4 tiles per block
    enc_gemm_kernel<0, 4><<<grid, 32, 0, stream>>>(X, ldx, xoff, Cin, idxb, KNB,
                                                   Wpk, ga, be, chsum, chsq,
                                                   out, 448, ooff, O, groups);
    enc_gemm_kernel<1, 4><<<grid, 32, 0, stream>>>(X, ldx, xoff, Cin, idxb, KNB,
                                                   Wpk, ga, be, chsum, chsq,
                                                   out, 448, ooff, O, groups);
  };

  // ---------------- encoder ----------------
  run_edge_layer(surface, 6, 0, 6,   pW1, g1, b1, cat, 0,   64,  8);
  run_edge_layer(cat, 448, 0, 64,    pW2, g2, b2, cat, 64,  128, 8);
  run_edge_layer(cat, 448, 64, 128,  pW3, g3, b3, cat, 192, 256, 16);

  auto run_proj = [&](const __bf16* Wpk, const float* ga, const float* be, float* out) {
    hipMemsetAsync(chsum, 0, sizeof(float) * BB * 256, stream);
    hipMemsetAsync(chsq,  0, sizeof(float) * BB * 256, stream);
    int grid = BB * NTILE * (256 / 64);
    enc_gemm_kernel<0, 4><<<grid, 32, 0, stream>>>(cat, 448, 0, 448, nullptr, 1,
                                                   Wpk, ga, be, chsum, chsq,
                                                   out, 256, 0, 256, 16);
    enc_gemm_kernel<1, 4><<<grid, 32, 0, stream>>>(cat, 448, 0, 448, nullptr, 1,
                                                   Wpk, ga, be, chsum, chsq,
                                                   out, 256, 0, 256, 16);
  };
  run_proj(pWp, gp, bp, pf);
  run_proj(pWg, gg, bg, gb);
  greduce_kernel<<<BB * 256, 256, 0, stream>>>(gb, gf);

  // ---------------- decoder ----------------
  sqnorm_kernel<<<(EV + 255) / 256, 256, 0, stream>>>(surface, 6, 0, 3, ssq, EV);
  {
    size_t n = (size_t)BB * TT * 3;
    copy_kernel<<<(unsigned)((n + 255) / 256), 256, 0, stream>>>(templ, pos, n);
  }

  const int MT = MDEC / 16;
  for (int step = 0; step < 3; ++step) {
    local_feat_kernel<<<BB * TT, 32, 0, stream>>>(pos, surface, ssq, pf, local);
    {
      size_t n = (size_t)BB * TT * NIP;
      build_ni_kernel<<<(unsigned)((n + 255) / 256), 256, 0, stream>>>(pos, local, gf, ni);
    }
    // h1 = relu(ni @ fc1^T + b)          O=256 -> 4 tile-groups
    dec_gemm_kernel<__bf16, 4><<<MT * 4, 32, 0, stream>>>(ni, NIP, pFc1, fc1b, h1, 256,
                                                          NIP, 256, 1, 0, 1.f);
    // h2 = ni @ skip^T + b
    dec_gemm_kernel<__bf16, 4><<<MT * 4, 32, 0, stream>>>(ni, NIP, pSk, skb, h2, 256,
                                                          NIP, 256, 0, 0, 1.f);
    // h2 += relu(h1 @ fc2^T + b)
    dec_gemm_kernel<__bf16, 4><<<MT * 4, 32, 0, stream>>>(h1, 256, pFc2, fc2b, h2, 256,
                                                          256, 256, 1, 1, 1.f);
    // disp = (h2 @ out^T + b) * 0.3   (f32 out, O=3)
    dec_gemm_kernel<float, 1><<<MT * 1, 32, 0, stream>>>(h2, 256, pOut, outb, disp, 3,
                                                         256, 3, 0, 0, 0.3f);
    // mat branch
    dec_gemm_kernel<__bf16, 4><<<MT * 2, 32, 0, stream>>>(ni, NIP, pM1, m1b, mb1, 128,
                                                          NIP, 128, 1, 0, 1.f);
    dec_gemm_kernel<__bf16, 4><<<MT * 1, 32, 0, stream>>>(mb1, 128, pM2, m2b, mb2, 64,
                                                          128, 64, 1, 0, 1.f);
    dec_gemm_kernel<float, 1><<<MT * 1, 32, 0, stream>>>(mb2, 64, pM3, m3b, out_mat, 1,
                                                         64, 1, 2, 0, 1.f);
    // pos += disp
    {
      size_t n = (size_t)BB * TT * 3;
      add_kernel<<<(unsigned)((n + 255) / 256), 256, 0, stream>>>(pos, disp, n);
    }
  }

  // out_delta = pos - template
  {
    size_t n = (size_t)BB * TT * 3;
    sub_kernel<<<(unsigned)((n + 255) / 256), 256, 0, stream>>>(pos, templ, out_delta, n);
  }
}
